// DiscSeqRNNDecoder_42073499632088
// MI455X (gfx1250) — matrix operand order repaired and checked by reading, not compiled, not run on hardware
//
#include <hip/hip_runtime.h>
#include <hip/hip_bf16.h>
#include <stdint.h>

// dims
#define B_DIM   8192
#define L_DIM   32
#define V_DIM   256
#define DIN_DIM 256
#define H_DIM   512
#define E_DIM   256

typedef __bf16 v16bf __attribute__((ext_vector_type(16)));
typedef float  v8f   __attribute__((ext_vector_type(8)));

union FragBF {
    uint4 q[2];
    unsigned short s[16];
    v16bf v;
};

union BFBits { __bf16 b; unsigned short s; };

__device__ __forceinline__ unsigned short f2bf(float f) {
    BFBits x; x.b = (__bf16)f;           // hardware v_cvt on gfx1250
    return x.s;
}

__device__ __forceinline__ float bf2f(unsigned short s) {
    union { unsigned u; float f; } x; x.u = ((unsigned)s) << 16;
    return x.f;
}

__device__ __forceinline__ float sigmoidf_(float x) {
    return 1.0f / (1.0f + __expf(-x));
}
__device__ __forceinline__ float tanhf_(float x) {
    float e = __expf(2.0f * x);
    return (e - 1.0f) / (e + 1.0f);
}

__device__ __forceinline__ v8f wmma_bf16(v16bf a, v16bf b, v8f c) {
    return __builtin_amdgcn_wmma_f32_16x16x32_bf16(
        /*neg_a=*/false, a, /*neg_b=*/false, b,
        /*c_mod=*/(short)0, c, /*reuse_a=*/false, /*reuse_b=*/false);
}

// A-fragment (16x32 bf16) from an LDS row-major bf16 buffer.
// ISA layout: lane<16: halves0-7 = K[kc..kc+7], halves8-15 = K[kc+16..kc+23]
//             lane>=16: +8 on the K base.
__device__ __forceinline__ v16bf load_a_lds(const unsigned short* buf, int ld,
                                            int kc, int lane) {
    int m  = lane & 15;
    int kb = kc + ((lane >> 4) << 3);
    const unsigned short* p = buf + m * ld + kb;
    FragBF f;
    f.q[0] = *(const uint4*)p;           // K = kb .. kb+7
    f.q[1] = *(const uint4*)(p + 16);    // K = kb+16 .. kb+23
    return f.v;
}

// A-fragment (16x32 bf16) from a global row-major bf16 matrix (x_bf).
__device__ __forceinline__ v16bf load_a_gl(const unsigned short* __restrict__ X,
                                           int row0, int kc, int ld, int lane) {
    int m  = row0 + (lane & 15);
    int kb = kc + ((lane >> 4) << 3);
    const unsigned short* p = X + (size_t)m * ld + kb;
    FragBF f;
    f.q[0] = *(const uint4*)p;
    f.q[1] = *(const uint4*)(p + 16);
    return f.v;
}

// B-fragment (32x16 bf16) from row-major global weight W[N][K] (B[k][n]=W[n][k]).
// ISA layout: lane<16: halves0-7 = K[kc..kc+7], halves8-15 = K[kc+16..kc+23]; lane>=16: +8 on the K base.
__device__ __forceinline__ v16bf load_b_gl(const unsigned short* __restrict__ W,
                                           int n0, int kc, int ld, int lane) {
    int n  = n0 + (lane & 15);
    int kb = kc + ((lane >> 4) << 3);
    const unsigned short* p = W + (size_t)n * ld + kb;
    FragBF f;
    f.q[0] = *(const uint4*)p;           // K = kb .. kb+7
    f.q[1] = *(const uint4*)(p + 16);    // K = kb+16 .. kb+23
    return f.v;
}

__global__ void cvt_bf16_kernel(const float* __restrict__ src,
                                unsigned short* __restrict__ dst, int n) {
    int i = blockIdx.x * blockDim.x + threadIdx.x;
    if (i < n) dst[i] = f2bf(src[i]);
}

// LDS row strides (bank-conflict padding)
#define LDH 520   // h_buf: 512 + 8 halves
#define LDI 264   // i_buf: 256 + 8 halves
#define LDL 257   // logits: 256 + 1 floats

__global__ __launch_bounds__(512, 1)
void rnn_decode_kernel(const unsigned short* __restrict__ x_bf,
                       const unsigned short* __restrict__ emb_bf,
                       const float* __restrict__ sos,
                       const unsigned short* __restrict__ Wi2h_bf,
                       const float* __restrict__ b_i2h,
                       const unsigned short* __restrict__ Wih_bf,
                       const unsigned short* __restrict__ Whh_bf,
                       const float* __restrict__ b_ih,
                       const float* __restrict__ b_hh,
                       const unsigned short* __restrict__ Wout_bf,
                       const float* __restrict__ b_out,
                       float* __restrict__ out_tok,
                       float* __restrict__ out_logits)
{
    __shared__ __align__(16) unsigned short h_buf[16 * LDH];  // h state, bf16
    __shared__ __align__(16) unsigned short i_buf[16 * LDI];  // input embed, bf16
    __shared__ __align__(16) float          lg_buf[16 * LDL]; // logits, fp32
    __shared__ int tok_buf[16];

    const int tid  = threadIdx.x;
    const int lane = tid & 31;
    const int wid  = tid >> 5;            // 0..15 waves
    const int m0   = blockIdx.x * 16;     // global batch-row base of this WG
    const int nb   = wid * 32;            // H-column slice base for this wave
    const int lm   = lane & 15;
    const int rsel = (lane >> 4) * 8;     // C-tile row offset for upper half-wave

    // ---- seed i_buf with sos_embed (broadcast across rows) ----
    {
        int m = tid >> 5;
        int e = (tid & 31) * 8;
#pragma unroll
        for (int j = 0; j < 8; ++j) i_buf[m * LDI + e + j] = f2bf(sos[e + j]);
    }

    // ---- h0 = x @ W_i2h^T + b_i2h  (wave computes cols [nb, nb+32)) ----
    {
        v8f c0 = {}; v8f c1 = {};
        for (int kc = 0; kc < DIN_DIM; kc += 32) {
            v16bf a = load_a_gl(x_bf, m0, kc, DIN_DIM, lane);
            c0 = wmma_bf16(a, load_b_gl(Wi2h_bf, nb,      kc, DIN_DIM, lane), c0);
            c1 = wmma_bf16(a, load_b_gl(Wi2h_bf, nb + 16, kc, DIN_DIM, lane), c1);
        }
        float bb0 = b_i2h[nb + lm];
        float bb1 = b_i2h[nb + 16 + lm];
#pragma unroll
        for (int r = 0; r < 8; ++r) {
            int m = r + rsel;
            h_buf[m * LDH + nb + lm]      = f2bf(c0[r] + bb0);
            h_buf[m * LDH + nb + 16 + lm] = f2bf(c1[r] + bb1);
        }
    }
    __syncthreads();

    // ---- sequence loop ----
    for (int step = 0; step < L_DIM; ++step) {
        // accumulators: r,z gates fuse gi+gh; n gate keeps i_n and h_n separate
        v8f arz0 = {}, arz1 = {}, azz0 = {}, azz1 = {};
        v8f ain0 = {}, ain1 = {}, ahn0 = {}, ahn1 = {};

        // gi = i @ W_ih^T  (K = E = 256)
        for (int kc = 0; kc < E_DIM; kc += 32) {
            v16bf a = load_a_lds(i_buf, LDI, kc, lane);
            arz0 = wmma_bf16(a, load_b_gl(Wih_bf, nb,                kc, E_DIM, lane), arz0);
            arz1 = wmma_bf16(a, load_b_gl(Wih_bf, nb + 16,           kc, E_DIM, lane), arz1);
            azz0 = wmma_bf16(a, load_b_gl(Wih_bf, H_DIM + nb,        kc, E_DIM, lane), azz0);
            azz1 = wmma_bf16(a, load_b_gl(Wih_bf, H_DIM + nb + 16,   kc, E_DIM, lane), azz1);
            ain0 = wmma_bf16(a, load_b_gl(Wih_bf, 2*H_DIM + nb,      kc, E_DIM, lane), ain0);
            ain1 = wmma_bf16(a, load_b_gl(Wih_bf, 2*H_DIM + nb + 16, kc, E_DIM, lane), ain1);
        }
        // gh = h @ W_hh^T  (K = H = 512)
        for (int kc = 0; kc < H_DIM; kc += 32) {
            v16bf a = load_a_lds(h_buf, LDH, kc, lane);
            arz0 = wmma_bf16(a, load_b_gl(Whh_bf, nb,                kc, H_DIM, lane), arz0);
            arz1 = wmma_bf16(a, load_b_gl(Whh_bf, nb + 16,           kc, H_DIM, lane), arz1);
            azz0 = wmma_bf16(a, load_b_gl(Whh_bf, H_DIM + nb,        kc, H_DIM, lane), azz0);
            azz1 = wmma_bf16(a, load_b_gl(Whh_bf, H_DIM + nb + 16,   kc, H_DIM, lane), azz1);
            ahn0 = wmma_bf16(a, load_b_gl(Whh_bf, 2*H_DIM + nb,      kc, H_DIM, lane), ahn0);
            ahn1 = wmma_bf16(a, load_b_gl(Whh_bf, 2*H_DIM + nb + 16, kc, H_DIM, lane), ahn1);
        }
        __syncthreads();   // all waves finished reading h_buf

        // ---- gate math + h update (each (m,n) owned by exactly one lane) ----
        {
            int n0 = nb + lm, n1 = nb + 16 + lm;
            float br0  = b_ih[n0] + b_hh[n0];
            float br1  = b_ih[n1] + b_hh[n1];
            float bz0  = b_ih[H_DIM + n0] + b_hh[H_DIM + n0];
            float bz1  = b_ih[H_DIM + n1] + b_hh[H_DIM + n1];
            float bin0 = b_ih[2*H_DIM + n0], bin1 = b_ih[2*H_DIM + n1];
            float bhn0 = b_hh[2*H_DIM + n0], bhn1 = b_hh[2*H_DIM + n1];
#pragma unroll
            for (int r = 0; r < 8; ++r) {
                int m = r + rsel;
                {
                    float rg   = sigmoidf_(arz0[r] + br0);
                    float zg   = sigmoidf_(azz0[r] + bz0);
                    float ng   = tanhf_(ain0[r] + bin0 + rg * (ahn0[r] + bhn0));
                    float hold = bf2f(h_buf[m * LDH + n0]);
                    h_buf[m * LDH + n0] = f2bf((1.0f - zg) * ng + zg * hold);
                }
                {
                    float rg   = sigmoidf_(arz1[r] + br1);
                    float zg   = sigmoidf_(azz1[r] + bz1);
                    float ng   = tanhf_(ain1[r] + bin1 + rg * (ahn1[r] + bhn1));
                    float hold = bf2f(h_buf[m * LDH + n1]);
                    h_buf[m * LDH + n1] = f2bf((1.0f - zg) * ng + zg * hold);
                }
            }
        }
        __syncthreads();   // new h visible to all waves

        // ---- logits = h_new @ W_out^T + b_out  (wave owns V-tile wid*16) ----
        {
            v8f cl = {};
            int nv = wid * 16;
            for (int kc = 0; kc < H_DIM; kc += 32) {
                v16bf a = load_a_lds(h_buf, LDH, kc, lane);
                cl = wmma_bf16(a, load_b_gl(Wout_bf, nv, kc, H_DIM, lane), cl);
            }
            float bo = b_out[nv + lm];
#pragma unroll
            for (int r = 0; r < 8; ++r) {
                int m = r + rsel;
                float v = cl[r] + bo;
                lg_buf[m * LDL + nv + lm] = v;
                out_logits[((size_t)(m0 + m) * L_DIM + step) * V_DIM + nv + lm] = v;
            }
        }
        __syncthreads();

        // ---- argmax over V=256: wave w reduces row w (first-max wins) ----
        {
            int m = wid;
            const float* row = &lg_buf[m * LDL];
            float best = row[lane * 8];
            int bi = lane * 8;
#pragma unroll
            for (int j = 1; j < 8; ++j) {
                int v = lane * 8 + j;
                float val = row[v];
                if (val > best) { best = val; bi = v; }
            }
#pragma unroll
            for (int off = 16; off >= 1; off >>= 1) {
                float ov = __shfl_xor(best, off, 32);
                int   oi = __shfl_xor(bi,  off, 32);
                if (ov > best || (ov == best && oi < bi)) { best = ov; bi = oi; }
            }
            if (lane == 0) {
                tok_buf[m] = bi;
                out_tok[(size_t)(m0 + m) * L_DIM + step] = (float)bi;
            }
        }
        __syncthreads();

        // ---- embedding gather: i_buf[m] = emb[tok[m]] ----
        {
            int m = tid >> 5;
            int e = (tid & 31) * 8;
            int t = tok_buf[m];
            *(uint4*)&i_buf[m * LDI + e] =
                *(const uint4*)(emb_bf + (size_t)t * E_DIM + e);
        }
        __syncthreads();
    }
}

// workspace layout (bf16 elements)
#define OFF_WIH  0
#define N_WIH    (3 * H_DIM * E_DIM)     // 1536*256
#define OFF_WHH  (OFF_WIH + N_WIH)
#define N_WHH    (3 * H_DIM * H_DIM)     // 1536*512
#define OFF_WOUT (OFF_WHH + N_WHH)
#define N_WOUT   (V_DIM * H_DIM)         // 256*512
#define OFF_EMB  (OFF_WOUT + N_WOUT)
#define N_EMB    (V_DIM * E_DIM)         // 256*256
#define OFF_WI2H (OFF_EMB + N_EMB)
#define N_WI2H   (H_DIM * DIN_DIM)       // 512*256
#define OFF_X    (OFF_WI2H + N_WI2H)
#define N_X      (B_DIM * DIN_DIM)       // 8192*256

extern "C" void kernel_launch(void* const* d_in, const int* in_sizes, int n_in,
                              void* d_out, int out_size, void* d_ws, size_t ws_size,
                              hipStream_t stream) {
    // setup_inputs order:
    // 0:x 1:emb 2:sos_embed 3:W_i2h 4:b_i2h 5:W_ih 6:W_hh 7:b_ih 8:b_hh 9:W_out 10:b_out
    const float* x     = (const float*)d_in[0];
    const float* emb   = (const float*)d_in[1];
    const float* sos   = (const float*)d_in[2];
    const float* Wi2h  = (const float*)d_in[3];
    const float* b_i2h = (const float*)d_in[4];
    const float* Wih   = (const float*)d_in[5];
    const float* Whh   = (const float*)d_in[6];
    const float* b_ih  = (const float*)d_in[7];
    const float* b_hh  = (const float*)d_in[8];
    const float* Wout  = (const float*)d_in[9];
    const float* b_out = (const float*)d_in[10];

    unsigned short* ws = (unsigned short*)d_ws;

    // fp32 -> bf16 conversion (deterministic, every call)
    auto cvt = [&](const float* src, unsigned short* dst, int n) {
        cvt_bf16_kernel<<<(n + 255) / 256, 256, 0, stream>>>(src, dst, n);
    };
    cvt(Wih,  ws + OFF_WIH,  N_WIH);
    cvt(Whh,  ws + OFF_WHH,  N_WHH);
    cvt(Wout, ws + OFF_WOUT, N_WOUT);
    cvt(emb,  ws + OFF_EMB,  N_EMB);
    cvt(Wi2h, ws + OFF_WI2H, N_WI2H);
    cvt(x,    ws + OFF_X,    N_X);

    float* out_tok    = (float*)d_out;
    float* out_logits = (float*)d_out + (size_t)B_DIM * L_DIM;

    rnn_decode_kernel<<<B_DIM / 16, 512, 0, stream>>>(
        ws + OFF_X, ws + OFF_EMB, sos, ws + OFF_WI2H, b_i2h,
        ws + OFF_WIH, ws + OFF_WHH, b_ih, b_hh,
        ws + OFF_WOUT, b_out, out_tok, out_logits);
}
